// SelfAttention_14860586844639
// MI455X (gfx1250) — compile-verified, matrix operand order checked by reading, not run on hardware
//
#include <hip/hip_runtime.h>

// ---------------------------------------------------------------------------
// Self-attention (DIM=1024, 16 heads, head_dim=64, B=2, N=2048) for gfx1250.
// All matmuls run on v_wmma_f32_16x16x32_bf16 (fp32 inputs cast to bf16,
// fp32 accumulation).  Working set (~56MB) is L2-resident (192MB), so GEMM
// fragments are loaded directly from global memory as b128 chunks.
// Round 2: bigger per-wave tiles (GEMM 32x128, attention 32 query rows) to
// raise FLOP/byte above the L2 roofline, plus prefetch of the next K tile.
// ---------------------------------------------------------------------------

typedef __attribute__((ext_vector_type(16))) __bf16 bfrag16;  // 8 VGPRs
typedef __attribute__((ext_vector_type(8)))  float  v8f;      // 8 VGPRs

constexpr int BB   = 2;
constexpr int SEQ  = 2048;
constexpr int CDIM = 1024;
constexpr int H    = 16;
constexpr int HD   = 64;
constexpr int TOK  = BB * SEQ;     // 4096 tokens
constexpr int F3   = 3 * CDIM;     // 3072
constexpr float ATTN_SCALE = 0.125f;  // 64^-0.5

// ---- helpers --------------------------------------------------------------

__device__ __forceinline__ __bf16 f2bf(float f) {
  unsigned int u = __float_as_uint(f);
  u += 0x7FFFu + ((u >> 16) & 1u);          // round-to-nearest-even
  unsigned short hs = (unsigned short)(u >> 16);
  __bf16 r;
  __builtin_memcpy(&r, &hs, sizeof(r));
  return r;
}

// Load a 16x32 bf16 WMMA fragment (A-layout; identical pattern serves as
// B-fragment when the 16 "rows" are the 16 output columns) from a row-major
// matrix with leading dimension `ld` (elements).  Per ISA 16-bit A layout:
//   lanes 0-15 : row = lane,    K chunks {0..7} and {16..23}
//   lanes 16-31: row = lane-16, K chunks {8..15} and {24..31}
__device__ __forceinline__ bfrag16 load_frag(const __bf16* p, int ld) {
  const int lane = threadIdx.x & 31;
  const int r  = lane & 15;
  const int hh = lane >> 4;
  const __bf16* q = p + (long)r * ld + hh * 8;
  uint4 lo = *(const uint4*)(q);        // K = hh*8 .. hh*8+7
  uint4 hi = *(const uint4*)(q + 16);   // K = hh*8+16 .. hh*8+23
  bfrag16 f;
  __builtin_memcpy(&f, &lo, 16);
  __builtin_memcpy(((char*)&f) + 16, &hi, 16);
  return f;
}

__device__ __forceinline__ v8f wmma_bf16(bfrag16 a, bfrag16 b, v8f c) {
  return __builtin_amdgcn_wmma_f32_16x16x32_bf16(
      /*neg_a=*/false, a, /*neg_b=*/false, b,
      /*c_mod=*/(short)0, c, /*reuse_a=*/false, /*reuse_b=*/false);
}

// ---- stage 1: fp32 -> bf16 cast ------------------------------------------

__global__ void cast_f32_bf16(const float* __restrict__ src,
                              __bf16* __restrict__ dst, int n) {
  int i = blockIdx.x * blockDim.x + threadIdx.x;
  if (i < n) dst[i] = f2bf(src[i]);
}

// ---- stage 2: qkv = x @ Wqkv^T  (bf16 out, plus V transposed copy) -------
// A: [TOK][CDIM] bf16, W: [F3][CDIM] bf16, Q: [TOK][F3] bf16,
// vT: [BB][H][HD][SEQ] bf16.
// Block = 256 threads = 8 waves arranged 4(M) x 2(N); wave tile 32x128.

__global__ __launch_bounds__(256)
void qkv_gemm(const __bf16* __restrict__ A, const __bf16* __restrict__ W,
              __bf16* __restrict__ Q, __bf16* __restrict__ vT) {
  const int wave = threadIdx.x >> 5;
  const int wm = wave >> 1, wn = wave & 1;
  const int m0 = blockIdx.y * 128 + wm * 32;
  const int n0 = blockIdx.x * 256 + wn * 128;

  v8f acc[2][8] = {};
  for (int k = 0; k < CDIM; k += 32) {
    bfrag16 a0 = load_frag(A + (long)m0 * CDIM + k, CDIM);
    bfrag16 a1 = load_frag(A + (long)(m0 + 16) * CDIM + k, CDIM);
#pragma unroll
    for (int j = 0; j < 8; ++j) {
      bfrag16 b = load_frag(W + (long)(n0 + j * 16) * CDIM + k, CDIM);
      acc[0][j] = wmma_bf16(a0, b, acc[0][j]);
      acc[1][j] = wmma_bf16(a1, b, acc[1][j]);
    }
  }

  const int lane  = threadIdx.x & 31;
  const int col   = lane & 15;
  const int rbase = (lane >> 4) * 8;
#pragma unroll
  for (int i = 0; i < 2; ++i) {
#pragma unroll
    for (int j = 0; j < 8; ++j) {
      const int f = n0 + j * 16 + col;
#pragma unroll
      for (int g = 0; g < 8; ++g) {
        const int row = m0 + i * 16 + rbase + g;    // token index
        __bf16 v = f2bf(acc[i][j][g]);
        Q[(long)row * F3 + f] = v;
        if (f >= 2 * CDIM) {                        // V part -> transposed copy
          const int rel = f - 2 * CDIM;
          const int hh = rel >> 6, dd = rel & 63;
          const int b  = row >> 11, n = row & (SEQ - 1);
          vT[(((long)b * H + hh) * HD + dd) * SEQ + n] = v;
        }
      }
    }
  }
}

// ---- stage 3: flash attention per (b, h, 128-query tile) -----------------
// qkv: [BB][SEQ][F3] bf16 (Q at +0, K at +CDIM, V at +2*CDIM per head slice)
// vT : [BB][H][HD][SEQ] bf16, attn out: [BB][SEQ][CDIM] bf16.
// Block = 128 threads = 4 waves; each wave owns 32 query rows (2 A tiles),
// doubling K/V fragment reuse vs 16 rows.

__global__ __launch_bounds__(128)
void attn_kernel(const __bf16* __restrict__ qkv, const __bf16* __restrict__ vT,
                 __bf16* __restrict__ attn) {
  const int w    = threadIdx.x >> 5;
  const int lane = threadIdx.x & 31;
  const int col   = lane & 15;
  const int rbase = (lane >> 4) * 8;

  const int qt = blockIdx.x;      // 0..15 (query tile of 128)
  const int h  = blockIdx.y;      // 0..15
  const int b  = blockIdx.z;      // 0..1
  const int m0 = qt * 128 + w * 32;

  // Q fragments: [query 16-row tile][d half]
  const __bf16* qbase = qkv + (long)(b * SEQ + m0) * F3 + h * HD;
  bfrag16 qa[2][2];
#pragma unroll
  for (int qi = 0; qi < 2; ++qi) {
    qa[qi][0] = load_frag(qbase + (long)qi * 16 * F3, F3);
    qa[qi][1] = load_frag(qbase + (long)qi * 16 * F3 + 32, F3);
  }

  const __bf16* kbase = qkv + (long)b * SEQ * F3 + CDIM + h * HD;
  const __bf16* vtb   = vT + ((long)(b * H + h)) * HD * SEQ;

  // Per-wave LDS tile for P (32 x 32 bf16, row stride 40 -> 80B, 16B aligned)
  __shared__ __bf16 plds_all[4][32 * 40];
  __bf16* plds = plds_all[w];

  float mrun[2][8], lrun[2][8];
#pragma unroll
  for (int qi = 0; qi < 2; ++qi)
#pragma unroll
    for (int g = 0; g < 8; ++g) { mrun[qi][g] = -1e30f; lrun[qi][g] = 0.f; }
  v8f o[2][4] = {};

  for (int n0 = 0; n0 < SEQ; n0 += 32) {
    // ---- load K fragments for 32 keys, compute S tiles ----
    bfrag16 kb[2][2];   // [key 16-tile][d half]
#pragma unroll
    for (int kt = 0; kt < 2; ++kt) {
      kb[kt][0] = load_frag(kbase + (long)(n0 + kt * 16) * F3, F3);
      kb[kt][1] = load_frag(kbase + (long)(n0 + kt * 16) * F3 + 32, F3);
    }
    if (n0 + 32 < SEQ) {  // prefetch next key tile (global_prefetch_b8)
      __builtin_prefetch(kbase + (long)(n0 + 32 + (lane & 15)) * F3, 0, 1);
      __builtin_prefetch(vtb + (long)(lane & 15) * SEQ + n0 + 32, 0, 1);
    }

    v8f s[2][2] = {};   // [query tile][key tile]
#pragma unroll
    for (int qi = 0; qi < 2; ++qi)
#pragma unroll
      for (int kt = 0; kt < 2; ++kt) {
        s[qi][kt] = wmma_bf16(qa[qi][0], kb[kt][0], s[qi][kt]);
        s[qi][kt] = wmma_bf16(qa[qi][1], kb[kt][1], s[qi][kt]);
      }

    // ---- online softmax per query tile (rows live in 16-lane halves) ----
#pragma unroll
    for (int qi = 0; qi < 2; ++qi) {
      float mnew[8], alpha[8];
#pragma unroll
      for (int g = 0; g < 8; ++g) {
        float v0 = s[qi][0][g] * ATTN_SCALE;
        float v1 = s[qi][1][g] * ATTN_SCALE;
        s[qi][0][g] = v0; s[qi][1][g] = v1;
        float mm = fmaxf(v0, v1);
        mm = fmaxf(mm, __shfl_xor(mm, 1));
        mm = fmaxf(mm, __shfl_xor(mm, 2));
        mm = fmaxf(mm, __shfl_xor(mm, 4));
        mm = fmaxf(mm, __shfl_xor(mm, 8));
        mnew[g]  = fmaxf(mrun[qi][g], mm);
        alpha[g] = __expf(mrun[qi][g] - mnew[g]);
        mrun[qi][g] = mnew[g];
      }
#pragma unroll
      for (int g = 0; g < 8; ++g) {
        float p0 = __expf(s[qi][0][g] - mnew[g]);
        float p1 = __expf(s[qi][1][g] - mnew[g]);
        float rs = p0 + p1;
        rs += __shfl_xor(rs, 1);
        rs += __shfl_xor(rs, 2);
        rs += __shfl_xor(rs, 4);
        rs += __shfl_xor(rs, 8);
        lrun[qi][g] = lrun[qi][g] * alpha[g] + rs;
        const int row = qi * 16 + rbase + g;
        plds[row * 40 + col]      = f2bf(p0);   // keys n0 .. n0+15
        plds[row * 40 + 16 + col] = f2bf(p1);   // keys n0+16 .. n0+31
      }
#pragma unroll
      for (int t = 0; t < 4; ++t)
#pragma unroll
        for (int g = 0; g < 8; ++g) o[qi][t][g] *= alpha[g];
    }

    asm volatile("s_wait_dscnt 0" ::: "memory");  // P visible to this wave

    // ---- O += P V (A = P from LDS, B = V columns from vT) ----
    bfrag16 pa[2];
    pa[0] = load_frag(plds, 40);
    pa[1] = load_frag(plds + 16 * 40, 40);
#pragma unroll
    for (int t = 0; t < 4; ++t) {
      bfrag16 vb = load_frag(vtb + (long)(t * 16) * SEQ + n0, SEQ);
      o[0][t] = wmma_bf16(pa[0], vb, o[0][t]);
      o[1][t] = wmma_bf16(pa[1], vb, o[1][t]);
    }
  }

  // ---- normalize and store bf16 ----
#pragma unroll
  for (int qi = 0; qi < 2; ++qi) {
    float inv[8];
#pragma unroll
    for (int g = 0; g < 8; ++g) inv[g] = 1.f / lrun[qi][g];
    __bf16* ob = attn + (long)(b * SEQ + m0 + qi * 16) * CDIM + h * HD;
#pragma unroll
    for (int t = 0; t < 4; ++t)
#pragma unroll
      for (int g = 0; g < 8; ++g)
        ob[(long)(rbase + g) * CDIM + t * 16 + col] = f2bf(o[qi][t][g] * inv[g]);
  }
}

// ---- stage 4: out = attn @ Wout^T + b_out (fp32 out) ---------------------

__global__ __launch_bounds__(256)
void out_gemm(const __bf16* __restrict__ A, const __bf16* __restrict__ W,
              const float* __restrict__ bias, float* __restrict__ Cf) {
  const int wave = threadIdx.x >> 5;
  const int wm = wave >> 1, wn = wave & 1;
  const int m0 = blockIdx.y * 128 + wm * 32;
  const int n0 = blockIdx.x * 256 + wn * 128;

  v8f acc[2][8] = {};
  for (int k = 0; k < CDIM; k += 32) {
    bfrag16 a0 = load_frag(A + (long)m0 * CDIM + k, CDIM);
    bfrag16 a1 = load_frag(A + (long)(m0 + 16) * CDIM + k, CDIM);
#pragma unroll
    for (int j = 0; j < 8; ++j) {
      bfrag16 b = load_frag(W + (long)(n0 + j * 16) * CDIM + k, CDIM);
      acc[0][j] = wmma_bf16(a0, b, acc[0][j]);
      acc[1][j] = wmma_bf16(a1, b, acc[1][j]);
    }
  }

  const int lane  = threadIdx.x & 31;
  const int col   = lane & 15;
  const int rbase = (lane >> 4) * 8;
#pragma unroll
  for (int i = 0; i < 2; ++i) {
#pragma unroll
    for (int j = 0; j < 8; ++j) {
      const int f = n0 + j * 16 + col;
      const float bv = bias[f];
#pragma unroll
      for (int g = 0; g < 8; ++g) {
        const int row = m0 + i * 16 + rbase + g;
        Cf[(long)row * CDIM + f] = acc[i][j][g] + bv;
      }
    }
  }
}

// ---- launcher -------------------------------------------------------------

extern "C" void kernel_launch(void* const* d_in, const int* in_sizes, int n_in,
                              void* d_out, int out_size, void* d_ws, size_t ws_size,
                              hipStream_t stream) {
  const float* x     = (const float*)d_in[0];   // [2,2048,1024]
  const float* w_qkv = (const float*)d_in[1];   // [3072,1024]
  const float* w_out = (const float*)d_in[2];   // [1024,1024]
  const float* b_out = (const float*)d_in[3];   // [1024]
  float* out = (float*)d_out;

  char* ws = (char*)d_ws;
  size_t off = 0;
  auto take = [&](size_t elems) {
    void* p = ws + off;
    off = (off + elems * sizeof(__bf16) + 255) & ~(size_t)255;
    return (__bf16*)p;
  };
  __bf16* xb   = take((size_t)TOK * CDIM);   // x bf16
  __bf16* wqb  = take((size_t)F3 * CDIM);    // Wqkv bf16
  __bf16* wob  = take((size_t)CDIM * CDIM);  // Wout bf16
  __bf16* qkvb = take((size_t)TOK * F3);     // qkv result
  __bf16* vTb  = take((size_t)TOK * CDIM);   // V transposed [b][h][d][n]
  __bf16* attb = take((size_t)TOK * CDIM);   // attention output

  const int nx = TOK * CDIM, nq = F3 * CDIM, no = CDIM * CDIM;
  cast_f32_bf16<<<(nx + 255) / 256, 256, 0, stream>>>(x, xb, nx);
  cast_f32_bf16<<<(nq + 255) / 256, 256, 0, stream>>>(w_qkv, wqb, nq);
  cast_f32_bf16<<<(no + 255) / 256, 256, 0, stream>>>(w_out, wob, no);

  qkv_gemm<<<dim3(F3 / 256, TOK / 128), 256, 0, stream>>>(xb, wqb, qkvb, vTb);
  attn_kernel<<<dim3(SEQ / 128, H, BB), 128, 0, stream>>>(qkvb, vTb, attb);
  out_gemm<<<dim3(CDIM / 256, TOK / 128), 256, 0, stream>>>(attb, wob, b_out, out);
}
